// CrossAttentionSPPPViT_72009421685103
// MI455X (gfx1250) — matrix operand order repaired and checked by reading, not compiled, not run on hardware
//
#include <hip/hip_runtime.h>
#include <hip/hip_bf16.h>
#include <hip/hip_fp16.h>

typedef _Float16 h16;
typedef __attribute__((ext_vector_type(16))) _Float16 v16h;
typedef __attribute__((ext_vector_type(8)))  _Float16 v8h;
typedef __attribute__((ext_vector_type(8)))  float    v8f;
typedef __attribute__((ext_vector_type(4)))  int      v4i_t;

constexpr int kDepth = 12;
constexpr int kD     = 768;
constexpr int kH     = 12;
constexpr int kHD    = 64;
constexpr int kFFN   = 3072;
constexpr int kB     = 64;
constexpr int kLQ    = 577;
constexpr int kSK    = 64;
constexpr int kMQ    = kB * kLQ;   // 36928
constexpr int kMKV   = kB * kSK;   // 4096
constexpr float kEPS = 1e-5f;

#if defined(__has_builtin)
#if __has_builtin(__builtin_amdgcn_global_load_async_to_lds_b128)
#define USE_ASYNC_LDS 1
#endif
#endif

static __device__ inline v8f wmma_f16(v16h a, v16h b, v8f c) {
  // (neg_a, A, neg_b, B, c_mod, C, reuse_a, reuse_b)
  return __builtin_amdgcn_wmma_f32_16x16x32_f16(false, a, false, b, (short)0, c,
                                                false, false);
}

static __device__ inline v16h cat8(v8h lo, v8h hi) {
  return __builtin_shufflevector(lo, hi, 0, 1, 2, 3, 4, 5, 6, 7, 8, 9, 10, 11,
                                 12, 13, 14, 15);
}

// 16-byte global -> LDS copy. Async (ASYNCcnt-tracked) when the gfx1250
// builtin is available; synchronous fallback keeps identical control flow.
static __device__ inline void ld_g2l_b128(const h16* __restrict__ g,
                                          h16* __restrict__ l) {
#ifdef USE_ASYNC_LDS
  h16* gnc = const_cast<h16*>(g);
  __builtin_amdgcn_global_load_async_to_lds_b128(
      (__attribute__((address_space(1))) v4i_t*)gnc,
      (__attribute__((address_space(3))) v4i_t*)l, 0, 0);
#else
  *reinterpret_cast<uint4*>(l) = *reinterpret_cast<const uint4*>(g);
#endif
}

static __device__ inline void wait_async_le8() {
#ifdef USE_ASYNC_LDS
#if __has_builtin(__builtin_amdgcn_s_wait_asynccnt)
  __builtin_amdgcn_s_wait_asynccnt(8);
#else
  asm volatile("s_wait_asynccnt 0x8" ::: "memory");
#endif
#endif
}

static __device__ inline void wait_async_0() {
#ifdef USE_ASYNC_LDS
#if __has_builtin(__builtin_amdgcn_s_wait_asynccnt)
  __builtin_amdgcn_s_wait_asynccnt(0);
#else
  asm volatile("s_wait_asynccnt 0x0" ::: "memory");
#endif
#endif
}

// ---------------------------------------------------------------------------
// f32 -> f16 convert (weights, once per launch)
// ---------------------------------------------------------------------------
__global__ void cvt_f16_kernel(const float* __restrict__ in,
                               h16* __restrict__ out, size_t n) {
  size_t i = (size_t)blockIdx.x * blockDim.x + threadIdx.x;
  size_t stride = (size_t)gridDim.x * blockDim.x;
  for (; i < n; i += stride) out[i] = (h16)in[i];
}

// ---------------------------------------------------------------------------
// LayerNorm (f32 in) -> f16 out.  One row (768) per block of 256 threads.
// ---------------------------------------------------------------------------
__global__ __launch_bounds__(256) void ln_f16_kernel(
    const float* __restrict__ X, const float* __restrict__ g,
    const float* __restrict__ bta, h16* __restrict__ out) {
  __shared__ float red[256];
  const int row = blockIdx.x;
  const int tid = threadIdx.x;
  const float* x = X + (size_t)row * kD;

  float v[3];
#pragma unroll
  for (int i = 0; i < 3; ++i) v[i] = x[tid + i * 256];

  float s = v[0] + v[1] + v[2];
  red[tid] = s;
  __syncthreads();
  for (int k = 128; k > 0; k >>= 1) {
    if (tid < k) red[tid] += red[tid + k];
    __syncthreads();
  }
  const float mu = red[0] * (1.0f / kD);
  __syncthreads();

  float d[3];
  float sq = 0.f;
#pragma unroll
  for (int i = 0; i < 3; ++i) {
    d[i] = v[i] - mu;
    sq += d[i] * d[i];
  }
  red[tid] = sq;
  __syncthreads();
  for (int k = 128; k > 0; k >>= 1) {
    if (tid < k) red[tid] += red[tid + k];
    __syncthreads();
  }
  const float rs = rsqrtf(red[0] * (1.0f / kD) + kEPS);

  h16* o = out + (size_t)row * kD;
#pragma unroll
  for (int i = 0; i < 3; ++i) {
    int c = tid + i * 256;
    o[c] = (h16)(d[i] * rs * g[c] + bta[c]);
  }
}

// ---------------------------------------------------------------------------
// WMMA GEMM: C[M,N] = A[M,K] (f16) x W[N,K]^T (f16) + bias
// MODE 0: store f16          (QKV projections)
// MODE 1: GELU, store f16    (FFN1)
// MODE 2: f32 residual +=    (O-proj, FFN2)
// Block: 256 threads (8 waves, 4x2), tile 128x128x64.
// Double-buffered dynamic LDS; next-stage tiles prefetched with async
// global->LDS copies overlapped with the WMMA phase of the current stage.
// ---------------------------------------------------------------------------
constexpr int kBM = 128, kBN = 128, kBK = 64, kBKP = 72;
constexpr int kStageH16 = (kBM + kBN) * kBKP;                // 18432 halfwords
constexpr size_t kGemmLdsBytes = 2u * kStageH16 * sizeof(h16);  // 73728 B

template <int MODE>
__global__ __launch_bounds__(256) void gemm_wmma_kernel(
    const h16* __restrict__ A, const h16* __restrict__ W,
    const float* __restrict__ bias, void* __restrict__ Cout, int M, int N,
    int K) {
  extern __shared__ h16 smem[];

  const int tid   = threadIdx.x;
  const int bm    = blockIdx.y * kBM;
  const int bn    = blockIdx.x * kBN;
  const int w     = tid >> 5;
  const int lane  = tid & 31;
  const int wm    = w & 3;   // 0..3
  const int wn    = w >> 2;  // 0..1
  const int mrow  = lane & 15;
  const int khalf = lane >> 4;

  // per-thread staging addresses (8 x 16B chunks per stage: 4 for A, 4 for W)
  int srow[4], scol[4];
  size_t arow[4], brow[4];
#pragma unroll
  for (int i = 0; i < 4; ++i) {
    int c = tid + i * 256;  // 0..1023 chunks of 8 halfwords
    srow[i] = c >> 3;
    scol[i] = (c & 7) * 8;
    int ga = bm + srow[i];
    if (ga >= M) ga = M - 1;  // clamp: garbage lands only in discarded rows
    arow[i] = (size_t)ga * K;
    int gb = bn + srow[i];
    if (gb >= N) gb = N - 1;
    brow[i] = (size_t)gb * K;
  }

  auto stage_issue = [&](int buf, int k0) {
    h16* sA = smem + buf * kStageH16;
    h16* sB = sA + kBM * kBKP;
#pragma unroll
    for (int i = 0; i < 4; ++i) {
      ld_g2l_b128(A + arow[i] + k0 + scol[i], &sA[srow[i] * kBKP + scol[i]]);
      ld_g2l_b128(W + brow[i] + k0 + scol[i], &sB[srow[i] * kBKP + scol[i]]);
    }
  };

  v8f acc[2][4] = {};

  stage_issue(0, 0);
  int cur = 0;
  for (int k0 = 0; k0 < K; k0 += kBK) {
    const bool has_next = (k0 + kBK) < K;
    if (has_next) {
      stage_issue(cur ^ 1, k0 + kBK);
      wait_async_le8();  // previous 8 copies (current stage) complete
    } else {
      wait_async_0();
    }
    __syncthreads();

    const h16* sA = smem + cur * kStageH16;
    const h16* sB = sA + kBM * kBKP;
#pragma unroll
    for (int kk = 0; kk < kBK; kk += 32) {
      v16h af[2];
#pragma unroll
      for (int mi = 0; mi < 2; ++mi) {
        int ar = wm * 32 + mi * 16 + mrow;
        v8h lo =
            *reinterpret_cast<const v8h*>(&sA[ar * kBKP + kk + khalf * 8]);
        v8h hi = *reinterpret_cast<const v8h*>(
            &sA[ar * kBKP + kk + 16 + khalf * 8]);
        af[mi] = cat8(lo, hi);
      }
      v16h bf[4];
#pragma unroll
      for (int ni = 0; ni < 4; ++ni) {
        int br = wn * 64 + ni * 16 + mrow;  // output feature (column of C)
        int kb = khalf * 8;
        v8h lo = *reinterpret_cast<const v8h*>(&sB[br * kBKP + kk + kb]);
        v8h hi = *reinterpret_cast<const v8h*>(&sB[br * kBKP + kk + kb + 16]);
        bf[ni] = cat8(lo, hi);
      }
#pragma unroll
      for (int mi = 0; mi < 2; ++mi)
#pragma unroll
        for (int ni = 0; ni < 4; ++ni)
          acc[mi][ni] = wmma_f16(af[mi], bf[ni], acc[mi][ni]);
    }
    __syncthreads();  // all waves done reading before buffer is re-filled
    cur ^= 1;
  }

  // epilogue: C layout lane(0..15)=col, VGPR r = row r (lanes 0-15) / r+8
#pragma unroll
  for (int mi = 0; mi < 2; ++mi) {
#pragma unroll
    for (int ni = 0; ni < 4; ++ni) {
      int col = bn + wn * 64 + ni * 16 + mrow;
      float bv = bias[col];
#pragma unroll
      for (int r = 0; r < 8; ++r) {
        int row = bm + wm * 32 + mi * 16 + r + khalf * 8;
        if (row < M) {
          float v = acc[mi][ni][r] + bv;
          size_t idx = (size_t)row * N + col;
          if constexpr (MODE == 0) {
            reinterpret_cast<h16*>(Cout)[idx] = (h16)v;
          } else if constexpr (MODE == 1) {
            v = 0.5f * v * (1.0f + erff(v * 0.70710678118654752f));
            reinterpret_cast<h16*>(Cout)[idx] = (h16)v;
          } else {
            float* C = reinterpret_cast<float*>(Cout);
            C[idx] += v;
          }
        }
      }
    }
  }
}

// ---------------------------------------------------------------------------
// Fused attention: one block per (b, h, 64-query tile). 128 threads, 4 waves.
// scores = Q K^T * 1/8 -> softmax(SK=64) -> P V -> O (f16).
// ---------------------------------------------------------------------------
__global__ __launch_bounds__(128) void attn_wmma_kernel(
    const h16* __restrict__ Q, const h16* __restrict__ Kb,
    const h16* __restrict__ Vb, h16* __restrict__ O) {
  __shared__ h16 Ks[64 * 72];        // K tile [s][d]
  __shared__ h16 Vts[64 * 72];       // V^T tile [d][s]
  __shared__ h16 Ps[4 * 16 * 72];    // per-wave probabilities [m][s]

  const int b  = blockIdx.z;
  const int h  = blockIdx.y;
  const int qt = blockIdx.x;
  const int tid   = threadIdx.x;
  const int w     = tid >> 5;
  const int lane  = tid & 31;
  const int mrow  = lane & 15;
  const int khalf = lane >> 4;

  // stage K tile; stage V transposed
#pragma unroll
  for (int i = 0; i < 4; ++i) {
    int c = tid + i * 128;   // 0..511 chunks of 8 halfwords (64x64 tile)
    int r = c >> 3;
    int cc = (c & 7) * 8;
    size_t gsrc = ((size_t)(b * kSK + r)) * kD + h * kHD + cc;
    uint4 kv = *reinterpret_cast<const uint4*>(Kb + gsrc);
    *reinterpret_cast<uint4*>(&Ks[r * 72 + cc]) = kv;
    v8h vv = *reinterpret_cast<const v8h*>(Vb + gsrc);
#pragma unroll
    for (int j = 0; j < 8; ++j) Vts[(cc + j) * 72 + r] = vv[j];
  }
  __syncthreads();

  const int q  = qt * 64 + w * 16 + mrow;
  const int qc = q < kLQ ? q : (kLQ - 1);
  const size_t qbase = ((size_t)b * kLQ + qc) * kD + h * kHD;

  // ---- scores = Q K^T ----
  v8f sacc[4] = {};
#pragma unroll
  for (int kk = 0; kk < 64; kk += 32) {
    v8h lo = *reinterpret_cast<const v8h*>(Q + qbase + kk + khalf * 8);
    v8h hi = *reinterpret_cast<const v8h*>(Q + qbase + kk + 16 + khalf * 8);
    v16h af = cat8(lo, hi);
#pragma unroll
    for (int ni = 0; ni < 4; ++ni) {
      int s = ni * 16 + mrow;
      int kb = khalf * 8;
      v8h blo = *reinterpret_cast<const v8h*>(&Ks[s * 72 + kk + kb]);
      v8h bhi = *reinterpret_cast<const v8h*>(&Ks[s * 72 + kk + kb + 16]);
      sacc[ni] = wmma_f16(af, cat8(blo, bhi), sacc[ni]);
    }
  }

  // ---- softmax over SK=64 (cross-lane within each 16-lane half) ----
  const float scale = 0.125f;  // 1/sqrt(64)
  h16* myP = &Ps[w * 16 * 72];
#pragma unroll
  for (int r = 0; r < 8; ++r) {
    float v0 = sacc[0][r] * scale, v1 = sacc[1][r] * scale;
    float v2 = sacc[2][r] * scale, v3 = sacc[3][r] * scale;
    float m = fmaxf(fmaxf(v0, v1), fmaxf(v2, v3));
#pragma unroll
    for (int msk = 1; msk < 16; msk <<= 1) m = fmaxf(m, __shfl_xor(m, msk));
    float e0 = __expf(v0 - m), e1 = __expf(v1 - m);
    float e2 = __expf(v2 - m), e3 = __expf(v3 - m);
    float sum = e0 + e1 + e2 + e3;
#pragma unroll
    for (int msk = 1; msk < 16; msk <<= 1) sum += __shfl_xor(sum, msk);
    float inv = 1.0f / sum;
    int prow = (r + khalf * 8) * 72;
    myP[prow + 0 * 16 + mrow] = (h16)(e0 * inv);
    myP[prow + 1 * 16 + mrow] = (h16)(e1 * inv);
    myP[prow + 2 * 16 + mrow] = (h16)(e2 * inv);
    myP[prow + 3 * 16 + mrow] = (h16)(e3 * inv);
  }

  // ---- O = P V ----
  v8f oacc[4] = {};
#pragma unroll
  for (int kk = 0; kk < 64; kk += 32) {
    v8h lo = *reinterpret_cast<const v8h*>(&myP[mrow * 72 + kk + khalf * 8]);
    v8h hi =
        *reinterpret_cast<const v8h*>(&myP[mrow * 72 + kk + 16 + khalf * 8]);
    v16h af = cat8(lo, hi);
#pragma unroll
    for (int ni = 0; ni < 4; ++ni) {
      int dcol = ni * 16 + mrow;
      int kb = khalf * 8;
      v8h blo = *reinterpret_cast<const v8h*>(&Vts[dcol * 72 + kk + kb]);
      v8h bhi = *reinterpret_cast<const v8h*>(&Vts[dcol * 72 + kk + kb + 16]);
      oacc[ni] = wmma_f16(af, cat8(blo, bhi), oacc[ni]);
    }
  }

#pragma unroll
  for (int ni = 0; ni < 4; ++ni) {
#pragma unroll
    for (int r = 0; r < 8; ++r) {
      int qq = qt * 64 + w * 16 + r + khalf * 8;
      if (qq < kLQ) {
        O[((size_t)b * kLQ + qq) * kD + h * kHD + ni * 16 + mrow] =
            (h16)oacc[ni][r];
      }
    }
  }
}

// ---------------------------------------------------------------------------
// Host driver
// ---------------------------------------------------------------------------
extern "C" void kernel_launch(void* const* d_in, const int* in_sizes, int n_in,
                              void* d_out, int out_size, void* d_ws,
                              size_t ws_size, hipStream_t stream) {
  const float* query = (const float*)d_in[0];
  const float* keyv  = (const float*)d_in[1];
  const float* ln1q_g = (const float*)d_in[2];
  const float* ln1q_b = (const float*)d_in[3];
  const float* ln1k_g = (const float*)d_in[4];
  const float* ln1k_b = (const float*)d_in[5];
  const float* wq = (const float*)d_in[6];
  const float* bq = (const float*)d_in[7];
  const float* wk = (const float*)d_in[8];
  const float* bk = (const float*)d_in[9];
  const float* wv = (const float*)d_in[10];
  const float* bv = (const float*)d_in[11];
  const float* wo = (const float*)d_in[12];
  const float* bo = (const float*)d_in[13];
  const float* ln2_g = (const float*)d_in[14];
  const float* ln2_b = (const float*)d_in[15];
  const float* w1 = (const float*)d_in[16];
  const float* b1 = (const float*)d_in[17];
  const float* w2 = (const float*)d_in[18];
  const float* b2 = (const float*)d_in[19];

  // workspace layout (f16 elements)
  h16* ws = (h16*)d_ws;
  const size_t szW  = (size_t)kDepth * kD * kD;     // 7,077,888
  const size_t szW1 = (size_t)kDepth * kFFN * kD;   // 28,311,552
  size_t off = 0;
  h16* WQ16 = ws + off; off += szW;
  h16* WK16 = ws + off; off += szW;
  h16* WV16 = ws + off; off += szW;
  h16* WO16 = ws + off; off += szW;
  h16* W116 = ws + off; off += szW1;
  h16* W216 = ws + off; off += szW1;
  h16* XN16 = ws + off; off += (size_t)kMQ * kD;
  h16* KV16 = ws + off; off += (size_t)kMKV * kD;
  h16* Q16  = ws + off; off += (size_t)kMQ * kD;
  h16* K16  = ws + off; off += (size_t)kMKV * kD;
  h16* V16  = ws + off; off += (size_t)kMKV * kD;
  h16* O16  = ws + off; off += (size_t)kMQ * kD;
  h16* H16  = ws + off; off += (size_t)kMQ * kFFN;

  // one-time weight conversion
  cvt_f16_kernel<<<4096, 256, 0, stream>>>(wq, WQ16, szW);
  cvt_f16_kernel<<<4096, 256, 0, stream>>>(wk, WK16, szW);
  cvt_f16_kernel<<<4096, 256, 0, stream>>>(wv, WV16, szW);
  cvt_f16_kernel<<<4096, 256, 0, stream>>>(wo, WO16, szW);
  cvt_f16_kernel<<<4096, 256, 0, stream>>>(w1, W116, szW1);
  cvt_f16_kernel<<<4096, 256, 0, stream>>>(w2, W216, szW1);

  float* X = (float*)d_out;  // running residual stream
  (void)hipMemcpyAsync(X, query, sizeof(float) * (size_t)kMQ * kD,
                       hipMemcpyDeviceToDevice, stream);

  const dim3 blk256(256), blk128(128);
  const dim3 gProjQ(kD / 128, (kMQ + 127) / 128);    // (6, 289)
  const dim3 gProjKV(kD / 128, kMKV / 128);          // (6, 32)
  const dim3 gFFN1(kFFN / 128, (kMQ + 127) / 128);   // (24, 289)
  const dim3 gAttn((kLQ + 63) / 64, kH, kB);         // (10, 12, 64)

  for (int l = 0; l < kDepth; ++l) {
    // LN1 on query stream and key_value
    ln_f16_kernel<<<kMQ, blk256, 0, stream>>>(X, ln1q_g + l * kD,
                                              ln1q_b + l * kD, XN16);
    ln_f16_kernel<<<kMKV, blk256, 0, stream>>>(keyv, ln1k_g + l * kD,
                                               ln1k_b + l * kD, KV16);
    // Q, K, V projections
    gemm_wmma_kernel<0><<<gProjQ, blk256, kGemmLdsBytes, stream>>>(
        XN16, WQ16 + (size_t)l * kD * kD, bq + l * kD, Q16, kMQ, kD, kD);
    gemm_wmma_kernel<0><<<gProjKV, blk256, kGemmLdsBytes, stream>>>(
        KV16, WK16 + (size_t)l * kD * kD, bk + l * kD, K16, kMKV, kD, kD);
    gemm_wmma_kernel<0><<<gProjKV, blk256, kGemmLdsBytes, stream>>>(
        KV16, WV16 + (size_t)l * kD * kD, bv + l * kD, V16, kMKV, kD, kD);
    // fused attention
    attn_wmma_kernel<<<gAttn, blk128, 0, stream>>>(Q16, K16, V16, O16);
    // output projection + residual
    gemm_wmma_kernel<2><<<gProjQ, blk256, kGemmLdsBytes, stream>>>(
        O16, WO16 + (size_t)l * kD * kD, bo + l * kD, X, kMQ, kD, kD);
    // LN2 + FFN
    ln_f16_kernel<<<kMQ, blk256, 0, stream>>>(X, ln2_g + l * kD,
                                              ln2_b + l * kD, XN16);
    gemm_wmma_kernel<1><<<gFFN1, blk256, kGemmLdsBytes, stream>>>(
        XN16, W116 + (size_t)l * kFFN * kD, b1 + l * kFFN, H16, kMQ, kFFN, kD);
    gemm_wmma_kernel<2><<<gProjQ, blk256, kGemmLdsBytes, stream>>>(
        H16, W216 + (size_t)l * kD * kFFN, b2 + l * kD, X, kMQ, kD, kFFN);
  }
}
